// VisionTransformerV_87445534146616
// MI455X (gfx1250) — matrix operand order repaired and checked by reading, not compiled, not run on hardware
//
#include <hip/hip_runtime.h>
#include <hip/hip_bf16.h>

// ---------------------------------------------------------------------------
// ViT + modern-Hopfield forward for MI455X (gfx1250, wave32, WMMA bf16 path).
// Heavy GEMMs stream fp32 operands from HBM (memory-bound: ~0.92 GB total ->
// ~40us floor at 23.3 TB/s), convert to bf16 in registers (v_cvt_pk_bf16_f32)
// and accumulate f32 via v_wmma_f32_16x16x32_bf16. Each wave keeps 4
// independent accumulators (2 M-tiles x 2 N-tiles) so consecutive WMMAs never
// chain on the same D registers -> no §7.12.1 hazard NOPs, and A fragments are
// reused across two B tiles.
// ---------------------------------------------------------------------------

typedef __attribute__((ext_vector_type(16))) __bf16 v16bf;
typedef __attribute__((ext_vector_type(8)))  float  v8f;

#define B_SZ   32
#define C_IN   3
#define IMG    224
#define PATCHD 16
#define EMBED  128
#define NPATCH 196
#define NFLAT  25088
#define NMEM   1024
#define HEADS  4
#define NCLS   1000
#define BETA_C 1.0f

// ---------------- WMMA helpers (A/B: contiguous k = 16h + s; C/D per CDNA5 ISA 7.12.2) ---------------

__device__ __forceinline__ v8f wmma_bf16(v16bf a, v16bf b, v8f c) {
  // (neg_a, A, neg_b, B, c_mod, C, reuse_a, reuse_b) -- probe-confirmed sig.
  return __builtin_amdgcn_wmma_f32_16x16x32_bf16(false, a, false, b,
                                                 (short)0, c, false, false);
}

// A-matrix 16x32 (MxK), 16-bit layout:
//  lanes 0-15 : row M=lane,   elems 0..15 = K k0..k0+15
//  lanes16-31 : row M=lane-16,elems 0..15 = K k0+16..k0+31
__device__ __forceinline__ v16bf load_a_bf16(const float* __restrict__ src,
                                             long row_stride, int row_base, int k0) {
  int lane = threadIdx.x & 31;
  int half = lane >> 4;
  const float* p = src + (long)(row_base + (lane & 15)) * row_stride + k0 + half * 16;
  v16bf a;
#pragma unroll
  for (int e = 0; e < 8; ++e) a[e] = (__bf16)p[e];
#pragma unroll
  for (int e = 0; e < 8; ++e) a[8 + e] = (__bf16)p[8 + e];
  return a;
}

// B-matrix 32x16 (KxN), source laid out as [k][n] with stride k_stride:
//  lanes 0-15: col=lane, K k0..k0+15 ; lanes 16-31: col=lane-16, K k0+16..k0+31
__device__ __forceinline__ v16bf load_b_bf16(const float* __restrict__ src,
                                             long k_stride, int k0, int n0) {
  int lane = threadIdx.x & 31;
  const float* p = src + (long)(k0 + ((lane >> 4) << 4)) * k_stride + n0 + (lane & 15);
  v16bf b;
#pragma unroll
  for (int e = 0; e < 16; ++e) b[e] = (__bf16)p[(long)e * k_stride];
  return b;
}

// B-matrix 32x16 where the SOURCE is transposed: element (k,n) = src[n*stride + k].
// Per-lane reads are 16 contiguous fp32 (64B) -> good coalescing while streaming.
__device__ __forceinline__ v16bf load_bt_bf16(const float* __restrict__ src,
                                              long row_stride, int k0, int n0) {
  int lane = threadIdx.x & 31;
  const float* p = src + (long)(n0 + (lane & 15)) * row_stride + k0 + ((lane >> 4) << 4);
  v16bf b;
#pragma unroll
  for (int e = 0; e < 16; ++e) b[e] = (__bf16)p[e];
  return b;
}

__device__ __forceinline__ v16bf load_bt_bf16_clamp(const float* __restrict__ src,
                                                    long row_stride, int k0, int n0,
                                                    int nmax) {
  int lane = threadIdx.x & 31;
  int col = n0 + (lane & 15);
  if (col >= nmax) col = nmax - 1;   // garbage lanes masked at store
  const float* p = src + (long)col * row_stride + k0 + ((lane >> 4) << 4);
  v16bf b;
#pragma unroll
  for (int e = 0; e < 16; ++e) b[e] = (__bf16)p[e];
  return b;
}

// C/D 16x16 f32: VGPR v -> (M=v, N=lane) lanes 0-15 ; (M=v+8, N=lane-16) lanes 16-31
__device__ __forceinline__ void store_c_f32(float* __restrict__ dst, long row_stride,
                                            int row_base, int col_base,
                                            const v8f& c, float scale) {
  int lane = threadIdx.x & 31;
  int col  = col_base + (lane & 15);
  int rofs = (lane >> 4) * 8;
#pragma unroll
  for (int v = 0; v < 8; ++v)
    dst[(long)(row_base + rofs + v) * row_stride + col] = c[v] * scale;
}

// ---------------- Kernel 1: im2col (patch gather, fp32) ---------------------
// Apatch[m=b*196+p][k=c*256+r*16+cc] = x[b][c][pr*16+r][pc*16+cc]
__global__ void k_im2col(const float* __restrict__ x, float* __restrict__ ap) {
  long id = (long)blockIdx.x * blockDim.x + threadIdx.x;
  const long total = (long)B_SZ * NPATCH * 768;
  if (id >= total) return;
  int  k  = (int)(id % 768);
  long m  = id / 768;
  int  b  = (int)(m / NPATCH), p = (int)(m % NPATCH);
  int  pr = p / 14, pc = p % 14;
  int  c  = k >> 8, r = (k >> 4) & 15, cc = k & 15;
  long src = (((long)b * C_IN + c) * IMG + (pr * PATCHD + r)) * IMG + (pc * PATCHD + cc);
  ap[id] = x[src];
}

// ---------------- Kernel 2: patch-embed GEMM + bias + pos -> x5 -------------
// C[6272x128] = Apatch[6272x768] * W^T ; 392 m-tiles, wave = n-tile (8 = 128/16)
// Two interleaved accumulator chains (even/odd K) so WMMAs never back-to-back
// on the same accumulator.
__global__ void k_patch_gemm(const float* __restrict__ ap, const float* __restrict__ w,
                             const float* __restrict__ bias, const float* __restrict__ pos,
                             float* __restrict__ x5) {
  int mtile = blockIdx.x;
  int wave  = threadIdx.x >> 5;
  int lane  = threadIdx.x & 31;
  int n0    = wave * 16;
  v8f ca = {}, cb = {};
#pragma unroll 3
  for (int k0 = 0; k0 < 768; k0 += 64) {
    v16bf a0 = load_a_bf16(ap, 768, mtile * 16, k0);
    v16bf b0 = load_bt_bf16(w, 768, k0, n0);        // w is [E][768] -> B^T form
    v16bf a1 = load_a_bf16(ap, 768, mtile * 16, k0 + 32);
    v16bf b1 = load_bt_bf16(w, 768, k0 + 32, n0);
    ca = wmma_bf16(a0, b0, ca);
    cb = wmma_bf16(a1, b1, cb);
  }
  int col  = n0 + (lane & 15);
  int rofs = (lane >> 4) * 8;
#pragma unroll
  for (int v = 0; v < 8; ++v) {
    int gp   = mtile * 16 + rofs + v;
    int bimg = gp / NPATCH, p = gp % NPATCH;
    x5[(long)bimg * NFLAT + (long)p * EMBED + col] =
        ca[v] + cb[v] + bias[col] + pos[(long)p * EMBED + col];
  }
}

// ---------------- Kernel 3: Hopfield scores = beta * x5 . X^T ---------------
// grid = H*32 (pairs of m-tiles of N_MEM); 8 waves split K=25088 (784 chunks),
// 4 independent accumulators per wave (2 batch-tiles x 2 m-tiles), LDS-reduce.
__global__ void k_scores(const float* __restrict__ x5, const float* __restrict__ X,
                         float* __restrict__ s) {
  __shared__ float red[32 * 256];
  int h     = blockIdx.x >> 5;
  int npair = blockIdx.x & 31;
  int wave  = threadIdx.x >> 5;
  int lane  = threadIdx.x & 31;
  const float* Xh = X + (long)h * NMEM * NFLAT;
  int n0 = npair * 32;
  v8f c00 = {}, c10 = {}, c01 = {}, c11 = {};
  for (int kc = wave; kc < 784; kc += 8) {
    int k0 = kc * 32;
    __builtin_prefetch(Xh + (long)n0 * NFLAT + k0 + 8 * 32, 0, 1); // stream ahead
    v16bf a0 = load_a_bf16(x5, NFLAT, 0,  k0);
    v16bf a1 = load_a_bf16(x5, NFLAT, 16, k0);
    v16bf b0 = load_bt_bf16(Xh, NFLAT, k0, n0);       // X rows contiguous in n(=K)
    v16bf b1 = load_bt_bf16(Xh, NFLAT, k0, n0 + 16);
    c00 = wmma_bf16(a0, b0, c00);
    c10 = wmma_bf16(a1, b0, c10);
    c01 = wmma_bf16(a0, b1, c01);
    c11 = wmma_bf16(a1, b1, c11);
  }
#pragma unroll
  for (int v = 0; v < 8; ++v) {
    red[(wave * 4 + 0) * 256 + lane * 8 + v] = c00[v];
    red[(wave * 4 + 1) * 256 + lane * 8 + v] = c10[v];
    red[(wave * 4 + 2) * 256 + lane * 8 + v] = c01[v];
    red[(wave * 4 + 3) * 256 + lane * 8 + v] = c11[v];
  }
  __syncthreads();
  if (wave < 4) {   // wave w reduces fragment type w
    v8f acc = {};
#pragma unroll
    for (int w2 = 0; w2 < 8; ++w2)
#pragma unroll
      for (int v = 0; v < 8; ++v)
        acc[v] += red[(w2 * 4 + wave) * 256 + lane * 8 + v];
    int row_base = (wave & 1) * 16;
    int col_base = n0 + (wave >> 1) * 16;
    // scores layout [b][h][m]: row stride H*NMEM, head offset h*NMEM
    store_c_f32(s + (long)h * NMEM, (long)HEADS * NMEM, row_base, col_base, acc, BETA_C);
  }
}

// ---------------- Kernel 4: softmax over N_MEM (in place) -------------------
__global__ void k_softmax(float* __restrict__ s) {
  __shared__ float red[256];
  float* row = s + (long)blockIdx.x * NMEM;   // blockIdx = b*HEADS + h
  int t = threadIdx.x;
  float v[4], mx = -1e30f;
#pragma unroll
  for (int i = 0; i < 4; ++i) { v[i] = row[t + i * 256]; mx = fmaxf(mx, v[i]); }
  red[t] = mx; __syncthreads();
  for (int off = 128; off > 0; off >>= 1) {
    if (t < off) red[t] = fmaxf(red[t], red[t + off]);
    __syncthreads();
  }
  mx = red[0]; __syncthreads();
  float sum = 0.f;
#pragma unroll
  for (int i = 0; i < 4; ++i) { v[i] = __expf(v[i] - mx); sum += v[i]; }
  red[t] = sum; __syncthreads();
  for (int off = 128; off > 0; off >>= 1) {
    if (t < off) red[t] += red[t + off];
    __syncthreads();
  }
  float inv = 1.0f / red[0];
#pragma unroll
  for (int i = 0; i < 4; ++i) row[t + i * 256] = v[i] * inv;
}

// ---------------- Kernel 5: retrieval + head-mean -> x7 ---------------------
// x7[b][n] = 0.25 * sum_h sum_m probs[b][h][m] * X[h][m][n]
// Heads fused into accumulators: X read exactly once here. Each wave owns a
// PAIR of n-tiles (32 cols) with 4 independent accumulators. 98 blocks x 8 waves.
__global__ void k_retrieve(const float* __restrict__ probs, const float* __restrict__ X,
                           float* __restrict__ x7) {
  int wave = threadIdx.x >> 5;
  int tp   = blockIdx.x * 8 + wave;   // 0..783 tile-pairs
  int n0   = tp * 32;
  v8f c00 = {}, c10 = {}, c01 = {}, c11 = {};
  for (int h = 0; h < HEADS; ++h) {
    const float* Xh = X + (long)h * NMEM * NFLAT;
    const float* Ph = probs + (long)h * NMEM;   // rows = batch, stride H*NMEM
#pragma unroll 2
    for (int kc = 0; kc < 32; ++kc) {
      int k0 = kc * 32;
      __builtin_prefetch(Xh + ((long)k0 + 64) * NFLAT + n0, 0, 1);
      v16bf a0 = load_a_bf16(Ph, (long)HEADS * NMEM, 0,  k0);
      v16bf a1 = load_a_bf16(Ph, (long)HEADS * NMEM, 16, k0);
      v16bf b0 = load_b_bf16(Xh, NFLAT, k0, n0);        // [m][n] natural layout
      v16bf b1 = load_b_bf16(Xh, NFLAT, k0, n0 + 16);
      c00 = wmma_bf16(a0, b0, c00);
      c10 = wmma_bf16(a1, b0, c10);
      c01 = wmma_bf16(a0, b1, c01);
      c11 = wmma_bf16(a1, b1, c11);
    }
  }
  store_c_f32(x7, NFLAT, 0,  n0,      c00, 0.25f);
  store_c_f32(x7, NFLAT, 16, n0,      c10, 0.25f);
  store_c_f32(x7, NFLAT, 0,  n0 + 16, c01, 0.25f);
  store_c_f32(x7, NFLAT, 16, n0 + 16, c11, 0.25f);
}

// ---------------- Kernel 6: final FC (x7 . fc_w^T + fc_b) -------------------
// 32 blocks x (pair of n-tiles = 32 classes); 8 waves split K, 4 accumulators,
// LDS-reduce; edge tiles clamped/guarded (N=1000).
__global__ void k_fc(const float* __restrict__ x7, const float* __restrict__ fw,
                     const float* __restrict__ fb, float* __restrict__ out) {
  __shared__ float red[32 * 256];
  int npair = blockIdx.x;
  int wave  = threadIdx.x >> 5;
  int lane  = threadIdx.x & 31;
  int n0    = npair * 32;
  v8f c00 = {}, c10 = {}, c01 = {}, c11 = {};
  for (int kc = wave; kc < 784; kc += 8) {
    int k0 = kc * 32;
    __builtin_prefetch(fw + (long)n0 * NFLAT + k0 + 8 * 32, 0, 1);
    v16bf a0 = load_a_bf16(x7, NFLAT, 0,  k0);
    v16bf a1 = load_a_bf16(x7, NFLAT, 16, k0);
    v16bf b0 = load_bt_bf16_clamp(fw, NFLAT, k0, n0,      NCLS);
    v16bf b1 = load_bt_bf16_clamp(fw, NFLAT, k0, n0 + 16, NCLS);
    c00 = wmma_bf16(a0, b0, c00);
    c10 = wmma_bf16(a1, b0, c10);
    c01 = wmma_bf16(a0, b1, c01);
    c11 = wmma_bf16(a1, b1, c11);
  }
#pragma unroll
  for (int v = 0; v < 8; ++v) {
    red[(wave * 4 + 0) * 256 + lane * 8 + v] = c00[v];
    red[(wave * 4 + 1) * 256 + lane * 8 + v] = c10[v];
    red[(wave * 4 + 2) * 256 + lane * 8 + v] = c01[v];
    red[(wave * 4 + 3) * 256 + lane * 8 + v] = c11[v];
  }
  __syncthreads();
  if (wave < 4) {
    v8f acc = {};
#pragma unroll
    for (int w2 = 0; w2 < 8; ++w2)
#pragma unroll
      for (int v = 0; v < 8; ++v)
        acc[v] += red[(w2 * 4 + wave) * 256 + lane * 8 + v];
    int row_base = (wave & 1) * 16;
    int col = n0 + (wave >> 1) * 16 + (lane & 15);
    if (col < NCLS) {
      int rofs = (lane >> 4) * 8;
#pragma unroll
      for (int v = 0; v < 8; ++v)
        out[(long)(row_base + rofs + v) * NCLS + col] = acc[v] + fb[col];
    }
  }
}

// ---------------------------------------------------------------------------

extern "C" void kernel_launch(void* const* d_in, const int* in_sizes, int n_in,
                              void* d_out, int out_size, void* d_ws, size_t ws_size,
                              hipStream_t stream) {
  const float* x      = (const float*)d_in[0];
  const float* conv_w = (const float*)d_in[1];
  const float* conv_b = (const float*)d_in[2];
  const float* pos    = (const float*)d_in[3];
  const float* hop_X  = (const float*)d_in[4];
  const float* fc_w   = (const float*)d_in[5];
  const float* fc_b   = (const float*)d_in[6];
  float* out = (float*)d_out;

  // Workspace layout (fp32 everywhere; bf16 conversion is done in-register).
  char* ws = (char*)d_ws;
  float* ap     = (float*)ws;                                     // 6272*768   = 19.3 MB
  float* x5     = (float*)(ws + (size_t)6272 * 768 * 4);          // 32*25088   =  3.2 MB
  float* scores = (float*)((char*)x5 + (size_t)B_SZ * NFLAT * 4); // 32*4*1024  =  0.5 MB
  float* x7     = (float*)((char*)scores + (size_t)B_SZ * HEADS * NMEM * 4); // 3.2 MB

  const long im2col_total = (long)B_SZ * NPATCH * 768;
  k_im2col<<<dim3((unsigned)((im2col_total + 255) / 256)), dim3(256), 0, stream>>>(x, ap);
  k_patch_gemm<<<dim3(392), dim3(256), 0, stream>>>(ap, conv_w, conv_b, pos, x5);
  k_scores<<<dim3(HEADS * 32), dim3(256), 0, stream>>>(x5, hop_X, scores);
  k_softmax<<<dim3(B_SZ * HEADS), dim3(256), 0, stream>>>(scores);
  k_retrieve<<<dim3(98), dim3(256), 0, stream>>>(scores, hop_X, x7);
  k_fc<<<dim3(32), dim3(256), 0, stream>>>(x7, fc_w, fc_b, out);
  (void)in_sizes; (void)n_in; (void)out_size; (void)ws_size;
}
